// ConvFrameMaskDecoderModular_9113920602357
// MI455X (gfx1250) — matrix operand order repaired and checked by reading, not compiled, not run on hardware
//
#include <hip/hip_runtime.h>

// ---------------------------------------------------------------------------
// CDNA5 / gfx1250 implementation of ConvFrameMaskDecoderModular forward.
//   Phase 0: pack weights to bf16; B-matrices packed into contiguous-k WMMA
//            fragment tiles (32B contiguous per lane per k-step).
//   Phase 1: vis_enc over all 400 frames as 3 big bf16 WMMA GEMMs (<2,2>).
//   Phase 2: 50-step scan; gate + ctrl GEMMs via v_wmma_f32_16x16x32_bf16
//            with <1,2> register blocking (no wasted row-tiles at M=8).
//   Phase 3: mask decoder batched over all 400 frames (chunked).
// K padded to multiples of 32 (1636->1664, 2148->2176) to kill guards.
// ---------------------------------------------------------------------------

typedef __bf16 bf16;
typedef __attribute__((ext_vector_type(16))) __bf16 v16bf;
typedef __attribute__((ext_vector_type(8)))  __bf16 v8bf;
typedef __attribute__((ext_vector_type(8)))  float  v8f;

#define DEV static __device__ __forceinline__

DEV float sigf(float x) { return 1.0f / (1.0f + expf(-x)); }

// --------------------------- small utility kernels -------------------------

__global__ void k_f2bf(const float* __restrict__ in, bf16* __restrict__ out, long n) {
  long i = (long)blockIdx.x * blockDim.x + threadIdx.x;
  if (i < n) out[i] = (bf16)in[i];
}

// in: (R, Ck) row-major f32  ->  out: (Ck, R) row-major bf16 (transpose)
__global__ void k_t2bf(const float* __restrict__ in, bf16* __restrict__ out, int R, int Ck) {
  long i = (long)blockIdx.x * blockDim.x + threadIdx.x;
  long n = (long)R * Ck;
  if (i >= n) return;
  int r = (int)(i / Ck), c = (int)(i % Ck);
  out[(long)c * R + r] = (bf16)in[i];
}

__global__ void k_vadd(const float* __restrict__ a, const float* __restrict__ b,
                       float* __restrict__ o, int n) {
  int i = blockIdx.x * blockDim.x + threadIdx.x;
  if (i < n) o[i] = a[i] + b[i];
}

// Wg[m][k][n] = k<1124 ? cell_wih[m][n][k] : cell_whh[m][n][k-1124]; (M,1636,2048)
__global__ void k_wg_pack(const float* __restrict__ wih, const float* __restrict__ whh,
                          bf16* __restrict__ Wg) {
  long i = (long)blockIdx.x * blockDim.x + threadIdx.x;
  long n = 8L * 1636 * 2048;
  if (i >= n) return;
  int g = (int)(i % 2048);
  long r = i / 2048;
  int k = (int)(r % 1636);
  int m = (int)(r / 1636);
  float v = (k < 1124) ? wih[((long)m * 2048 + g) * 1124 + k]
                       : whh[((long)m * 2048 + g) * 512 + (k - 1124)];
  Wg[i] = (bf16)v;
}

// Wc[k][n] = k<1636 ? ctrl_wih[n][k] : ctrl_whh[n][k-1636]; (2148,2048)
__global__ void k_wc_pack(const float* __restrict__ wih, const float* __restrict__ whh,
                          bf16* __restrict__ Wc) {
  long i = (long)blockIdx.x * blockDim.x + threadIdx.x;
  long n = 2148L * 2048;
  if (i >= n) return;
  int g = (int)(i % 2048);
  int k = (int)(i / 2048);
  float v = (k < 1636) ? wih[(long)g * 1636 + k] : whh[(long)g * 512 + (k - 1636)];
  Wc[i] = (bf16)v;
}

// Pack B (K,N) row-major bf16 -> WMMA fragment tiles, K zero-padded to Kp.
// dst[((kt*(N/16)+nt)*32 + lane)*16 + i] = B[kt*32 + (lane>>4)*16 + i][nt*16+(lane&15)]
__global__ void k_bpack(const bf16* __restrict__ src, bf16* __restrict__ dst,
                        int K, int N, int Kp, int nz, long srcZ) {
  long total = (long)nz * Kp * N;
  long idx = (long)blockIdx.x * blockDim.x + threadIdx.x;
  if (idx >= total) return;
  int i = (int)(idx & 15);
  int lane = (int)((idx >> 4) & 31);
  long t = idx >> 9;
  int nT = N >> 4;
  int nt = (int)(t % nT); t /= nT;
  int kT = Kp >> 5;
  int kt = (int)(t % kT);
  int z  = (int)(t / kT);
  int k = kt * 32 + (lane >> 4) * 16 + i;
  int n = nt * 16 + (lane & 15);
  dst[idx] = (k < K) ? src[(long)z * srcZ + (long)k * N + n] : (bf16)0.0f;
}

// frames (BT,512,7,7) f32 -> P1 (BT*49, 512) bf16 (pixel-major rows)
__global__ void k_frames_pack(const float* __restrict__ frames, bf16* __restrict__ P1) {
  long i = (long)blockIdx.x * blockDim.x + threadIdx.x;
  long n = 19600L * 512;
  if (i >= n) return;
  int c = (int)(i % 512);
  long row = i / 512;
  int p = (int)(row % 49);
  long bt = row / 49;
  P1[i] = (bf16)frames[((long)bt * 512 + c) * 49 + p];
}

// G2f (BT*49, 64) f32 -> X3 (BT, 3136) bf16 with 3136 = c*49 + p
__global__ void k_g2pack(const float* __restrict__ G2, bf16* __restrict__ X3) {
  long i = (long)blockIdx.x * blockDim.x + threadIdx.x;
  long n = 400L * 3136;
  if (i >= n) return;
  int rem = (int)(i % 3136);
  long bt = i / 3136;
  int c = rem / 49, p = rem % 49;
  X3[i] = (bf16)G2[((long)bt * 49 + p) * 64 + c];
}

__global__ void k_init_e(const float* __restrict__ go, float* __restrict__ e) {
  int i = blockIdx.x * blockDim.x + threadIdx.x;
  if (i < 8 * 100) e[i] = go[i % 100];
}

// --------------------------- WMMA GEMM (bf16 -> f32) -----------------------
// C[z](M,N) = act( (A[z](M,lda) * Bp[z] + bias[z])*bnScale + bnShift )
// A: row-major bf16, lda multiple of 8 elements, valid K = Kp (zero padded).
// Bp: fragment-packed (see k_bpack). One wave per (16*MT)x(16*NT) tile block.
template <int MT, int NT>
__global__ void k_gemm_bf16(const bf16* __restrict__ A, const bf16* __restrict__ Bp,
                            const float* __restrict__ bias, long biasStride,
                            const float* __restrict__ bng, const float* __restrict__ bnb,
                            float* __restrict__ C, int M, int N, int Kp, int lda,
                            long strideA, long strideB, long strideC,
                            int relu, float bnMul) {
  int z = blockIdx.z;
  A  += (long)z * strideA;
  Bp += (long)z * strideB;
  C  += (long)z * strideC;
  const float* biasp = bias ? (bias + (long)z * biasStride) : nullptr;

  int m0 = blockIdx.y * (16 * MT);
  int n0 = blockIdx.x * (16 * NT);
  int lane = threadIdx.x & 31;
  int hi = lane >> 4;
  int lr = lane & 15;

  const bf16* aPtr[MT];
#pragma unroll
  for (int mt = 0; mt < MT; ++mt) {
    int arow = m0 + mt * 16 + lr;
    int ar = (arow < M) ? arow : (M - 1);   // clamp: bad rows discarded at store
    aPtr[mt] = A + (long)ar * lda + hi * 16;
  }
  const bf16* bPtr[NT];
#pragma unroll
  for (int nt = 0; nt < NT; ++nt)
    bPtr[nt] = Bp + (((long)blockIdx.x * NT + nt) * 32 + lane) * 16;
  const long bStep = (long)N * 32;

  v8f acc[MT][NT];
#pragma unroll
  for (int mt = 0; mt < MT; ++mt)
#pragma unroll
    for (int nt = 0; nt < NT; ++nt)
#pragma unroll
      for (int v = 0; v < 8; ++v) acc[mt][nt][v] = 0.f;

  int ktiles = Kp >> 5;
  for (int kt = 0; kt < ktiles; ++kt) {
    v16bf a[MT], b[NT];
#pragma unroll
    for (int mt = 0; mt < MT; ++mt) {
      v8bf a0 = *(const v8bf*)(aPtr[mt]);
      v8bf a1 = *(const v8bf*)(aPtr[mt] + 8);
      a[mt] = __builtin_shufflevector(a0, a1, 0, 1, 2, 3, 4, 5, 6, 7,
                                      8, 9, 10, 11, 12, 13, 14, 15);
    }
#pragma unroll
    for (int nt = 0; nt < NT; ++nt) {
      v8bf b0 = *(const v8bf*)(bPtr[nt]);
      v8bf b1 = *(const v8bf*)(bPtr[nt] + 8);
      b[nt] = __builtin_shufflevector(b0, b1, 0, 1, 2, 3, 4, 5, 6, 7,
                                      8, 9, 10, 11, 12, 13, 14, 15);
    }
    if (kt + 1 < ktiles) {
#pragma unroll
      for (int mt = 0; mt < MT; ++mt) __builtin_prefetch(aPtr[mt] + 32, 0, 1);
#pragma unroll
      for (int nt = 0; nt < NT; ++nt) __builtin_prefetch(bPtr[nt] + bStep, 0, 1);
    }
#pragma unroll
    for (int mt = 0; mt < MT; ++mt)
#pragma unroll
      for (int nt = 0; nt < NT; ++nt)
        acc[mt][nt] = __builtin_amdgcn_wmma_f32_16x16x32_bf16(
            false, a[mt], false, b[nt], (short)0, acc[mt][nt], false, false);
#pragma unroll
    for (int mt = 0; mt < MT; ++mt) aPtr[mt] += 32;
#pragma unroll
    for (int nt = 0; nt < NT; ++nt) bPtr[nt] += bStep;
  }

#pragma unroll
  for (int mt = 0; mt < MT; ++mt) {
#pragma unroll
    for (int nt = 0; nt < NT; ++nt) {
#pragma unroll
      for (int v = 0; v < 8; ++v) {
        int r = m0 + mt * 16 + v + 8 * hi;
        int cj = n0 + nt * 16 + lr;
        if (r < M) {
          float val = acc[mt][nt][v];
          if (biasp) val += biasp[cj];
          if (bng)   val = val * (bng[cj] * bnMul) + bnb[cj];
          if (relu)  val = fmaxf(val, 0.f);
          C[(long)r * N + cj] = val;
        }
      }
    }
  }
}

// --------------------------- per-step kernels ------------------------------

// Per (b,m): q = hfc_w[m]@h[b]+hfc_b[m]; scores = softmax(enc[b]@q);
// weighted = scores@enc[b]; assemble Ag[m][b] = [vis|weighted|e|h|0pad] (1664)
__global__ void k_attn(const float* __restrict__ enc, const float* __restrict__ hfc_w,
                       const float* __restrict__ hfc_b, const float* __restrict__ h,
                       const float* __restrict__ vis, const float* __restrict__ e,
                       float* __restrict__ scores, bf16* __restrict__ Ag,
                       int t, int T) {
  int b = blockIdx.x >> 3, m = blockIdx.x & 7;
  int tid = threadIdx.x;                 // 128 threads
  __shared__ float q[512];
  __shared__ float sl[125];
  __shared__ float red[128];

  const float* hb = h + b * 512;
  for (int d = tid; d < 512; d += 128) {
    const float* wr = hfc_w + ((long)m * 512 + d) * 512;
    float s = hfc_b[m * 512 + d];
    for (int j = 0; j < 512; ++j) s += wr[j] * hb[j];
    q[d] = s;
  }
  __syncthreads();
  const float* encb = enc + (long)b * 125 * 512;
  for (int l = tid; l < 125; l += 128) {
    const float* er = encb + (long)l * 512;
    float s = 0.f;
    for (int j = 0; j < 512; ++j) s += er[j] * q[j];
    sl[l] = s;
  }
  __syncthreads();
  float mx = -1e30f;
  for (int l = tid; l < 125; l += 128) mx = fmaxf(mx, sl[l]);
  red[tid] = mx; __syncthreads();
  for (int s = 64; s > 0; s >>= 1) { if (tid < s) red[tid] = fmaxf(red[tid], red[tid + s]); __syncthreads(); }
  mx = red[0]; __syncthreads();
  float sm = 0.f;
  for (int l = tid; l < 125; l += 128) { float ev = expf(sl[l] - mx); sl[l] = ev; sm += ev; }
  red[tid] = sm; __syncthreads();
  for (int s = 64; s > 0; s >>= 1) { if (tid < s) red[tid] += red[tid + s]; __syncthreads(); }
  float inv = 1.f / red[0]; __syncthreads();
  for (int l = tid; l < 125; l += 128) { sl[l] *= inv; scores[((long)b * 8 + m) * 125 + l] = sl[l]; }
  __syncthreads();

  bf16* row = Ag + ((long)m * 8 + b) * 1664;
  for (int d = tid; d < 512; d += 128) {
    float s = 0.f;
    for (int l = 0; l < 125; ++l) s += sl[l] * encb[(long)l * 512 + d];
    row[512 + d] = (bf16)s;
  }
  const float* visb = vis + ((long)b * T + t) * 512;
  for (int d = tid; d < 512; d += 128) row[d] = (bf16)visb[d];
  for (int d = tid; d < 100; d += 128) row[1024 + d] = (bf16)e[b * 100 + d];
  for (int d = tid; d < 512; d += 128) row[1124 + d] = (bf16)hb[d];
  for (int d = tid; d < 28; d += 128)  row[1636 + d] = (bf16)0.0f;   // K pad
}

// LSTM cell nonlinearity for 8 modules sharing (h,c) input state.
__global__ void k_cell(const float* __restrict__ gates, const float* __restrict__ c_prev,
                       float* __restrict__ h_t, float* __restrict__ c_t) {
  int idx = blockIdx.x * blockDim.x + threadIdx.x;   // B*M*512 = 32768
  if (idx >= 8 * 8 * 512) return;
  int d = idx & 511, m = (idx >> 9) & 7, b = idx >> 12;
  const float* g = gates + ((long)m * 8 + b) * 2048;
  float ig = g[d], fg = g[512 + d], gg = g[1024 + d], og = g[1536 + d];
  float cp = c_prev[b * 512 + d];
  float ct = sigf(fg) * cp + sigf(ig) * tanhf(gg);
  float ht = sigf(og) * tanhf(ct);
  long o = ((long)b * 8 + m) * 512 + d;
  c_t[o] = ct;
  h_t[o] = ht;
}

// Controller attention + state mix + actor head + greedy embedding feedback.
__global__ void k_ctrl_attn(const float* __restrict__ chfc_w, const float* __restrict__ chfc_b,
                            const float* __restrict__ ch, const float* __restrict__ h_t,
                            const float* __restrict__ c_t, const bf16* __restrict__ Ag,
                            const float* __restrict__ scores, const float* __restrict__ actor_w,
                            const float* __restrict__ actor_b, const float* __restrict__ emb,
                            float* __restrict__ h, float* __restrict__ c,
                            bf16* __restrict__ cont_bf, bf16* __restrict__ Ac,
                            float* __restrict__ e, float* __restrict__ outActions,
                            float* __restrict__ outAttns, float* __restrict__ outMattns,
                            int t, int T) {
  int b = blockIdx.x, tid = threadIdx.x;   // 256 threads
  __shared__ float chq[512];
  __shared__ float ms[8];
  __shared__ float contS[1636];
  __shared__ float tmp100[100];
  __shared__ float act[64];
  __shared__ int amax;

  const float* chb = ch + b * 512;
  for (int d = tid; d < 512; d += 256) {
    const float* wr = chfc_w + (long)d * 512;
    float s = chfc_b[d];
    for (int j = 0; j < 512; ++j) s += wr[j] * chb[j];
    chq[d] = s;
  }
  __syncthreads();
  if (tid < 8) {
    const float* hr = h_t + ((long)b * 8 + tid) * 512;
    float s = 0.f;
    for (int j = 0; j < 512; ++j) s += hr[j] * chq[j];
    ms[tid] = s;
  }
  __syncthreads();
  if (tid == 0) {
    float mx = ms[0]; for (int i = 1; i < 8; ++i) mx = fmaxf(mx, ms[i]);
    float sm = 0.f; for (int i = 0; i < 8; ++i) { ms[i] = expf(ms[i] - mx); sm += ms[i]; }
    for (int i = 0; i < 8; ++i) ms[i] /= sm;
  }
  __syncthreads();
  long bt = (long)b * T + t;
  if (tid < 8) outMattns[bt * 8 + tid] = ms[tid];
  for (int d = tid; d < 512; d += 256) {
    float hs = 0.f, cs = 0.f;
    for (int m = 0; m < 8; ++m) {
      long o = ((long)b * 8 + m) * 512 + d;
      hs += ms[m] * h_t[o];
      cs += ms[m] * c_t[o];
    }
    h[b * 512 + d] = hs;
    c[b * 512 + d] = cs;
    contS[d] = hs;
  }
  for (int k = tid; k < 1124; k += 256) {
    float s = 0.f;
    for (int m = 0; m < 8; ++m) s += ms[m] * (float)Ag[((long)m * 8 + b) * 1664 + k];
    contS[512 + k] = s;
  }
  for (int l = tid; l < 125; l += 256) {
    float s = 0.f;
    for (int m = 0; m < 8; ++m) s += ms[m] * scores[((long)b * 8 + m) * 125 + l];
    outAttns[bt * 125 + l] = s;
  }
  __syncthreads();
  for (int k = tid; k < 1664; k += 256) {
    float v = (k < 1636) ? contS[k] : 0.f;
    cont_bf[bt * 1664 + k] = (bf16)v;
  }
  for (int k = tid; k < 2176; k += 256) {
    float v = (k < 1636) ? contS[k] : (k < 2148 ? chb[k - 1636] : 0.f);
    Ac[(long)b * 2176 + k] = (bf16)v;
  }
  for (int p = tid; p < 100; p += 256) {
    const float* ar = actor_w + (long)p * 1636;
    float s = actor_b[p];
    for (int k = 0; k < 1636; ++k) s += ar[k] * contS[k];
    tmp100[p] = s;
  }
  __syncthreads();
  if (tid < 64) {
    const float* er = emb + (long)tid * 100;
    float s = 0.f;
    for (int p = 0; p < 100; ++p) s += er[p] * tmp100[p];
    act[tid] = s;
    outActions[bt * 64 + tid] = s;
  }
  __syncthreads();
  if (tid == 0) {
    int bi = 0; float bv = act[0];
    for (int i = 1; i < 64; ++i) if (act[i] > bv) { bv = act[i]; bi = i; }
    amax = bi;
  }
  __syncthreads();
  for (int p = tid; p < 100; p += 256) e[b * 100 + p] = emb[(long)amax * 100 + p];
}

__global__ void k_ctrl_cell(const float* __restrict__ cg, float* __restrict__ ch,
                            float* __restrict__ cc) {
  int idx = blockIdx.x * blockDim.x + threadIdx.x;   // B*512 = 4096
  if (idx >= 8 * 512) return;
  int d = idx & 511, b = idx >> 9;
  const float* g = cg + (long)b * 2048;
  float ci = g[d], cf = g[512 + d], cgg = g[1024 + d], co = g[1536 + d];
  float ccv = sigf(cf) * cc[idx] + sigf(ci) * tanhf(cgg);
  cc[idx] = ccv;
  ch[idx] = sigf(co) * tanhf(ccv);
}

// --------------------------- mask decoder kernels --------------------------

// ConvTranspose2d(k=4,s=2,p=1), optional folded 2x nearest upsample on input,
// optional BN(+ReLU). X: (NF,Ci,Hin,Hin) -> Y: (NF,Co,Hout,Hout)
__global__ void k_convT(const float* __restrict__ X, const float* __restrict__ W,
                        const float* __restrict__ bias, const float* __restrict__ bng,
                        const float* __restrict__ bnb, float* __restrict__ Y,
                        int NF, int Ci, int Co, int Hin, int up2, int relu, float bnMul) {
  extern __shared__ float sw[];
  int nW = Ci * Co * 16;
  for (int i = threadIdx.x; i < nW; i += blockDim.x) sw[i] = W[i];
  __syncthreads();
  int Hup = up2 ? Hin * 2 : Hin;
  int Hout = Hup * 2;
  long total = (long)NF * Co * Hout * Hout;
  long idx = (long)blockIdx.x * blockDim.x + threadIdx.x;
  if (idx >= total) return;
  int ox = (int)(idx % Hout); long r = idx / Hout;
  int oy = (int)(r % Hout);   r /= Hout;
  int o  = (int)(r % Co);
  int n  = (int)(r / Co);
  float acc = bias[o];
  for (int ky = 0; ky < 4; ++ky) {
    int ty = oy + 1 - ky;
    if (ty < 0 || (ty & 1)) continue;
    int iy = ty >> 1; if (iy >= Hup) continue;
    int ry = up2 ? (iy >> 1) : iy;
    for (int kx = 0; kx < 4; ++kx) {
      int tx = ox + 1 - kx;
      if (tx < 0 || (tx & 1)) continue;
      int ix = tx >> 1; if (ix >= Hup) continue;
      int rx = up2 ? (ix >> 1) : ix;
      const float* xb = X + (((long)n * Ci) * Hin + ry) * Hin + rx;
      const float* wb = sw + o * 16 + ky * 4 + kx;
      for (int i = 0; i < Ci; ++i)
        acc += wb[(long)i * Co * 16] * xb[(long)i * Hin * Hin];
    }
  }
  if (bng) acc = acc * (bng[o] * bnMul) + bnb[o];
  if (relu) acc = fmaxf(acc, 0.f);
  Y[idx] = acc;
}

// bilinear 224x224 -> 300x300 (half-pixel centers, edge clamp)
__global__ void k_resize(const float* __restrict__ X, float* __restrict__ masks,
                         int NF, int frame0) {
  long total = (long)NF * 300 * 300;
  long idx = (long)blockIdx.x * blockDim.x + threadIdx.x;
  if (idx >= total) return;
  int x = (int)(idx % 300); long r = idx / 300;
  int y = (int)(r % 300);
  int n = (int)(r / 300);
  const float sc = 224.0f / 300.0f;
  float sx = (x + 0.5f) * sc - 0.5f;
  float sy = (y + 0.5f) * sc - 0.5f;
  sx = fminf(fmaxf(sx, 0.f), 223.f);
  sy = fminf(fmaxf(sy, 0.f), 223.f);
  int x0 = (int)sx, y0 = (int)sy;
  int x1 = min(x0 + 1, 223), y1 = min(y0 + 1, 223);
  float fx = sx - x0, fy = sy - y0;
  const float* p = X + (long)n * 224 * 224;
  float v00 = p[y0 * 224 + x0], v01 = p[y0 * 224 + x1];
  float v10 = p[y1 * 224 + x0], v11 = p[y1 * 224 + x1];
  float v = (v00 * (1 - fx) + v01 * fx) * (1 - fy) + (v10 * (1 - fx) + v11 * fx) * fy;
  masks[(long)(frame0 + n) * 90000 + y * 300 + x] = v;
}

// --------------------------- host side -------------------------------------

static inline long ceilDiv(long a, long b) { return (a + b - 1) / b; }

extern "C" void kernel_launch(void* const* d_in, const int* in_sizes, int n_in,
                              void* d_out, int out_size, void* d_ws, size_t ws_size,
                              hipStream_t stream) {
  (void)in_sizes; (void)n_in; (void)out_size; (void)ws_size;
  const float* enc        = (const float*)d_in[0];
  const float* frames     = (const float*)d_in[1];
  const float* h0         = (const float*)d_in[2];
  const float* c0         = (const float*)d_in[3];
  const float* ch0        = (const float*)d_in[4];
  const float* cc0        = (const float*)d_in[5];
  const float* emb        = (const float*)d_in[6];
  const float* go         = (const float*)d_in[7];
  const float* vc1w       = (const float*)d_in[8];
  const float* vc1b       = (const float*)d_in[9];
  const float* vbn1g      = (const float*)d_in[10];
  const float* vbn1b      = (const float*)d_in[11];
  const float* vc2w       = (const float*)d_in[12];
  const float* vc2b       = (const float*)d_in[13];
  const float* vbn2g      = (const float*)d_in[14];
  const float* vbn2b      = (const float*)d_in[15];
  const float* vfcw       = (const float*)d_in[16];
  const float* vfcb       = (const float*)d_in[17];
  const float* hfc_w      = (const float*)d_in[18];
  const float* hfc_b      = (const float*)d_in[19];
  const float* cell_wih   = (const float*)d_in[20];
  const float* cell_whh   = (const float*)d_in[21];
  const float* cell_bih   = (const float*)d_in[22];
  const float* cell_bhh   = (const float*)d_in[23];
  const float* chfc_w     = (const float*)d_in[24];
  const float* chfc_b     = (const float*)d_in[25];
  const float* ctrl_wih   = (const float*)d_in[26];
  const float* ctrl_whh   = (const float*)d_in[27];
  const float* ctrl_bih   = (const float*)d_in[28];
  const float* ctrl_bhh   = (const float*)d_in[29];
  const float* actor_w    = (const float*)d_in[30];
  const float* actor_b    = (const float*)d_in[31];
  const float* md_d1_w    = (const float*)d_in[32];
  const float* md_d1_b    = (const float*)d_in[33];
  const float* md_dc3_w   = (const float*)d_in[34];
  const float* md_dc3_b   = (const float*)d_in[35];
  const float* md_bn2_g   = (const float*)d_in[36];
  const float* md_bn2_b   = (const float*)d_in[37];
  const float* md_dc2_w   = (const float*)d_in[38];
  const float* md_dc2_b   = (const float*)d_in[39];
  const float* md_bn1_g   = (const float*)d_in[40];
  const float* md_bn1_b   = (const float*)d_in[41];
  const float* md_dc1_w   = (const float*)d_in[42];
  const float* md_dc1_b   = (const float*)d_in[43];

  const int B = 8, T = 50, BT = 400;
  const float BN_MUL = 0.999995000037496f;   // 1/sqrt(1+1e-5)

  // workspace allocator
  char* wsb = (char*)d_ws;
  size_t off = 0;
  auto alloc = [&](size_t bytes) -> void* {
    void* p = wsb + off;
    off = (off + bytes + 255) & ~(size_t)255;
    return p;
  };
  bf16* P1     = (bf16*)alloc(19600L * 512 * 2);
  bf16* w1t    = (bf16*)alloc(512L * 256 * 2);
  bf16* w2t    = (bf16*)alloc(256L * 64 * 2);
  bf16* wfct   = (bf16*)alloc(3136L * 512 * 2);
  bf16* md1t   = (bf16*)alloc(1636L * 3136 * 2);
  bf16* Wg     = (bf16*)alloc(8L * 1636 * 2048 * 2);
  bf16* Wc     = (bf16*)alloc(2148L * 2048 * 2);
  bf16* w1tP   = (bf16*)alloc(512L * 256 * 2);
  bf16* w2tP   = (bf16*)alloc(256L * 64 * 2);
  bf16* wfctP  = (bf16*)alloc(3136L * 512 * 2);
  bf16* md1tP  = (bf16*)alloc(1664L * 3136 * 2);
  bf16* WgP    = (bf16*)alloc(8L * 1664 * 2048 * 2);
  bf16* WcP    = (bf16*)alloc(2176L * 2048 * 2);
  float* gbs   = (float*)alloc(8L * 2048 * 4);
  float* cbs   = (float*)alloc(2048L * 4);
  float* G1f   = (float*)alloc(19600L * 256 * 4);
  bf16* G1b    = (bf16*)alloc(19600L * 256 * 2);
  float* G2f   = (float*)alloc(19600L * 64 * 4);
  bf16* X3     = (bf16*)alloc(400L * 3136 * 2);
  float* vis   = (float*)alloc(400L * 512 * 4);
  float* hS    = (float*)alloc(8L * 512 * 4);
  float* cS    = (float*)alloc(8L * 512 * 4);
  float* chS   = (float*)alloc(8L * 512 * 4);
  float* ccS   = (float*)alloc(8L * 512 * 4);
  float* eS    = (float*)alloc(8L * 100 * 4);
  float* scB   = (float*)alloc(8L * 8 * 125 * 4);
  bf16* Ag     = (bf16*)alloc(8L * 8 * 1664 * 2);
  float* gates = (float*)alloc(8L * 8 * 2048 * 4);
  float* h_t   = (float*)alloc(8L * 8 * 512 * 4);
  float* c_t   = (float*)alloc(8L * 8 * 512 * 4);
  bf16* Ac     = (bf16*)alloc(8L * 2176 * 2);
  float* cgates = (float*)alloc(8L * 2048 * 4);
  bf16* contB  = (bf16*)alloc(400L * 1664 * 2);
  const int CH = 40;                             // mask-decoder chunk (frames)
  float* d1    = (float*)alloc((long)CH * 3136 * 4);
  float* t3    = (float*)alloc((long)CH * 32 * 28 * 28 * 4);
  float* t2    = (float*)alloc((long)CH * 16 * 112 * 112 * 4);
  float* t1    = (float*)alloc((long)CH * 224 * 224 * 4);

  float* outActions = (float*)d_out;                       // (B,T,64)
  float* outMasks   = outActions + 25600L;                 // (B,T,1,300,300)
  float* outAttns   = outMasks + 36000000L;                // (B,T,125)
  float* outMattns  = outAttns + 50000L;                   // (B,T,8)

  // MTsel=2: <2,2> 32x32 blocks (big GEMMs); MTsel=1: <1,2> 16x32 (M=8 path)
  auto gemm = [&](int MTsel, const bf16* A, const bf16* Bp, const float* bias,
                  long biasStride, const float* bng, const float* bnb, float* C,
                  int Mr, int Nc, int Kp, int lda, long sA, long sB, long sC, int nz,
                  int relu, float bnMul) {
    if (MTsel == 2) {
      dim3 g((unsigned)(Nc / 32), (unsigned)ceilDiv(Mr, 32), (unsigned)nz);
      k_gemm_bf16<2, 2><<<g, 32, 0, stream>>>(A, Bp, bias, biasStride, bng, bnb, C,
                                              Mr, Nc, Kp, lda, sA, sB, sC, relu, bnMul);
    } else {
      dim3 g((unsigned)(Nc / 32), (unsigned)ceilDiv(Mr, 16), (unsigned)nz);
      k_gemm_bf16<1, 2><<<g, 32, 0, stream>>>(A, Bp, bias, biasStride, bng, bnb, C,
                                              Mr, Nc, Kp, lda, sA, sB, sC, relu, bnMul);
    }
  };
  const int THR = 256;
  auto g1d = [&](long n) { return dim3((unsigned)ceilDiv(n, THR)); };

  // ---------------- Phase 0: weight packing -------------------------------
  k_t2bf<<<g1d(256L * 512), THR, 0, stream>>>(vc1w, w1t, 256, 512);
  k_t2bf<<<g1d(64L * 256), THR, 0, stream>>>(vc2w, w2t, 64, 256);
  k_t2bf<<<g1d(512L * 3136), THR, 0, stream>>>(vfcw, wfct, 512, 3136);
  k_t2bf<<<g1d(3136L * 1636), THR, 0, stream>>>(md_d1_w, md1t, 3136, 1636);
  k_wg_pack<<<g1d(8L * 1636 * 2048), THR, 0, stream>>>(cell_wih, cell_whh, Wg);
  k_wc_pack<<<g1d(2148L * 2048), THR, 0, stream>>>(ctrl_wih, ctrl_whh, Wc);
  k_vadd<<<g1d(8L * 2048), THR, 0, stream>>>(cell_bih, cell_bhh, gbs, 8 * 2048);
  k_vadd<<<g1d(2048), THR, 0, stream>>>(ctrl_bih, ctrl_bhh, cbs, 2048);
  // fragment-pack all B matrices
  k_bpack<<<g1d(512L * 256), THR, 0, stream>>>(w1t, w1tP, 512, 256, 512, 1, 0);
  k_bpack<<<g1d(256L * 64), THR, 0, stream>>>(w2t, w2tP, 256, 64, 256, 1, 0);
  k_bpack<<<g1d(3136L * 512), THR, 0, stream>>>(wfct, wfctP, 3136, 512, 3136, 1, 0);
  k_bpack<<<g1d(1664L * 3136), THR, 0, stream>>>(md1t, md1tP, 1636, 3136, 1664, 1, 0);
  k_bpack<<<g1d(8L * 1664 * 2048), THR, 0, stream>>>(Wg, WgP, 1636, 2048, 1664, 8,
                                                     1636L * 2048);
  k_bpack<<<g1d(2176L * 2048), THR, 0, stream>>>(Wc, WcP, 2148, 2048, 2176, 1, 0);

  // ---------------- Phase 1: vis_enc over all 400 frames ------------------
  k_frames_pack<<<g1d(19600L * 512), THR, 0, stream>>>(frames, P1);
  gemm(2, P1, w1tP, vc1b, 0, vbn1g, vbn1b, G1f, 19600, 256, 512, 512,
       0, 0, 0, 1, 1, BN_MUL);
  k_f2bf<<<g1d(19600L * 256), THR, 0, stream>>>(G1f, G1b, 19600L * 256);
  gemm(2, G1b, w2tP, vc2b, 0, vbn2g, vbn2b, G2f, 19600, 64, 256, 256,
       0, 0, 0, 1, 1, BN_MUL);
  k_g2pack<<<g1d(400L * 3136), THR, 0, stream>>>(G2f, X3);
  gemm(2, X3, wfctP, vfcb, 0, nullptr, nullptr, vis, 400, 512, 3136, 3136,
       0, 0, 0, 1, 0, 1.f);

  // ---------------- Phase 2: recurrent scan -------------------------------
  hipMemcpyAsync(hS, h0, 8 * 512 * 4, hipMemcpyDeviceToDevice, stream);
  hipMemcpyAsync(cS, c0, 8 * 512 * 4, hipMemcpyDeviceToDevice, stream);
  hipMemcpyAsync(chS, ch0, 8 * 512 * 4, hipMemcpyDeviceToDevice, stream);
  hipMemcpyAsync(ccS, cc0, 8 * 512 * 4, hipMemcpyDeviceToDevice, stream);
  k_init_e<<<g1d(800), THR, 0, stream>>>(go, eS);

  for (int t = 0; t < T; ++t) {
    k_attn<<<dim3(64), 128, 0, stream>>>(enc, hfc_w, hfc_b, hS, vis, eS, scB, Ag, t, T);
    gemm(1, Ag, WgP, gbs, 2048, nullptr, nullptr, gates, 8, 2048, 1664, 1664,
         8L * 1664, 1664L * 2048, 8L * 2048, 8, 0, 1.f);
    k_cell<<<g1d(8L * 8 * 512), THR, 0, stream>>>(gates, cS, h_t, c_t);
    k_ctrl_attn<<<dim3(B), 256, 0, stream>>>(chfc_w, chfc_b, chS, h_t, c_t, Ag, scB,
                                             actor_w, actor_b, emb, hS, cS, contB, Ac,
                                             eS, outActions, outAttns, outMattns, t, T);
    gemm(1, Ac, WcP, cbs, 0, nullptr, nullptr, cgates, 8, 2048, 2176, 2176,
         0, 0, 0, 1, 0, 1.f);
    k_ctrl_cell<<<g1d(8L * 512), THR, 0, stream>>>(cgates, chS, ccS);
  }

  // ---------------- Phase 3: batched mask decoder -------------------------
  for (int f0 = 0; f0 < BT; f0 += CH) {
    gemm(2, contB + (long)f0 * 1664, md1tP, md_d1_b, 0, nullptr, nullptr, d1,
         CH, 3136, 1664, 1664, 0, 0, 0, 1, 1, 1.f);
    long n3 = (long)CH * 32 * 28 * 28;
    k_convT<<<g1d(n3), THR, (size_t)(64 * 32 * 16 * 4), stream>>>(
        d1, md_dc3_w, md_dc3_b, md_bn2_g, md_bn2_b, t3, CH, 64, 32, 7, 1, 1, BN_MUL);
    long n2 = (long)CH * 16 * 112 * 112;
    k_convT<<<g1d(n2), THR, (size_t)(32 * 16 * 16 * 4), stream>>>(
        t3, md_dc2_w, md_dc2_b, md_bn1_g, md_bn1_b, t2, CH, 32, 16, 28, 1, 1, BN_MUL);
    long n1 = (long)CH * 1 * 224 * 224;
    k_convT<<<g1d(n1), THR, (size_t)(16 * 1 * 16 * 4), stream>>>(
        t2, md_dc1_w, md_dc1_b, nullptr, nullptr, t1, CH, 16, 1, 112, 0, 0, 1.f);
    k_resize<<<g1d((long)CH * 300 * 300), THR, 0, stream>>>(t1, outMasks, CH, f0);
  }
}
